// BuildNN_GNN_MTL_72129680769295
// MI455X (gfx1250) — matrix operand order repaired and checked by reading, not compiled, not run on hardware
//
#include <hip/hip_runtime.h>
#include <hip/hip_bf16.h>

typedef __bf16 v16bf __attribute__((ext_vector_type(16)));
typedef float  v8f   __attribute__((ext_vector_type(8)));

constexpr int kNodes  = 50000;
constexpr int kEdges  = 500000;
constexpr int kGraphs = 512;
constexpr int NF = 168, EF = 2, NN = 128, EN = 64;
constexpr int ZDIM = 2 * NN + EN;   // 320
constexpr int ZPAD = 328;           // row pitch in bf16 elems (mult of 8 -> 16B aligned chunks)
constexpr int KSTEPS = ZDIM / 32;   // 10
constexpr int kConvBlocks = 2048;   // grid-stride persistent blocks for the edge GEMM
constexpr float EPS = 1e-5f;

// ---------------------------------------------------------------- utilities
__global__ void k_zero(float* __restrict__ p, int n) {
  int i = blockIdx.x * 256 + threadIdx.x;
  if (i < n) p[i] = 0.f;
}

// ------------------------------------------------- node embed: h = x@W1 + b1
__global__ void __launch_bounds__(128)
k_node_embed(const float* __restrict__ x, const float* __restrict__ W,
             const float* __restrict__ b, float* __restrict__ h,
             __bf16* __restrict__ hbf) {
  __shared__ float xr[NF];
  const int row = blockIdx.x;
  const int c   = threadIdx.x;                       // 0..127
  for (int k = c; k < NF; k += 128) xr[k] = x[row * NF + k];
  __syncthreads();
  float acc = b[c];
#pragma unroll 4
  for (int k = 0; k < NF; ++k) acc = fmaf(xr[k], W[k * NN + c], acc);
  h[row * NN + c]   = acc;
  hbf[row * NN + c] = (__bf16)acc;
}

// -------------------- transpose+convert CGConv weight: Wt[n][k] = bf16(W[k][n])
__global__ void k_prep_wt(const float* __restrict__ W, __bf16* __restrict__ Wt) {
  int i = blockIdx.x * 256 + threadIdx.x;            // over ZDIM*NN
  if (i >= ZDIM * NN) return;
  int k = i / NN, n = i % NN;
  Wt[n * ZDIM + k] = (__bf16)W[k * NN + n];
}

// --------------------------------------------------------- CGConv edge GEMM
// Persistent blocks; 8 wave32 per block, wave w owns output cols [16w,16w+16).
// B fragments (both gate matrices, all 10 K-steps) are register-resident and
// amortized over ~(numTiles/gridDim) edge tiles per block. A fragments are
// software-pipelined from LDS so ds-load latency hides under the WMMA chain.
__global__ void __launch_bounds__(256)
k_cgconv(const __bf16* __restrict__ hbf,
         const int* __restrict__ ei,          // [2, E]
         const float* __restrict__ ea,        // [E, 2]
         const float* __restrict__ W2,        // [2, 64]
         const float* __restrict__ b2,        // [64]
         const __bf16* __restrict__ Wtf,      // [NN, ZDIM] bf16 (B^T)
         const float* __restrict__ bfv,       // [NN]
         const __bf16* __restrict__ Wts,      // [NN, ZDIM] bf16 (B^T)
         const float* __restrict__ bsv,       // [NN]
         float* __restrict__ acc) {           // [N, NN] scatter target
  __shared__ __align__(16) __bf16 zs[16 * ZPAD];
  __shared__ int dsts[16];

  const int tid  = threadIdx.x;
  const int eL   = tid >> 4;                   // staging: local edge 0..15
  const int part = tid & 15;                   // staging: 8-col chunk 0..15
  const int wave = tid >> 5;                   // compute: 0..7 -> col tile
  const int lane = tid & 31;
  const int half = lane >> 4;                  // K half selector
  const int lm   = lane & 15;                  // row (A) / col (B,D)
  const int col  = wave * 16 + lm;

  // ---- hoist B fragments into registers (per-wave, reused across all tiles)
  v16bf Bf[KSTEPS], Bs[KSTEPS];
  {
    const __bf16* wf_r = Wtf + col * ZDIM;
    const __bf16* ws_r = Wts + col * ZDIM;
#pragma unroll
    for (int s = 0; s < KSTEPS; ++s) {
      // B k order chosen here (matches A): lane group `half` holds a contiguous K span of 16
      Bf[s] = *(const v16bf*)(wf_r + s * 32 + half * 16);
      Bs[s] = *(const v16bf*)(ws_r + s * 32 + half * 16);
    }
  }
  const float bgate = bfv[col];
  const float bsoft = bsv[col];

  const int numTiles = kEdges / 16;
  for (int t = blockIdx.x; t < numTiles; t += gridDim.x) {
    __syncthreads();   // prev iteration's compute/atomics done before restaging

    // ---- stage z = [h[dst] | h[src] | e_edge] as bf16 into LDS
    {
      const int eg   = t * 16 + eL;
      const int srcN = ei[eg];
      const int dstN = ei[kEdges + eg];
      if (part == 0) dsts[eL] = dstN;
      *(uint4*)(zs + eL * ZPAD + part * 8) =
          *(const uint4*)(hbf + (size_t)dstN * NN + part * 8);
      *(uint4*)(zs + eL * ZPAD + NN + part * 8) =
          *(const uint4*)(hbf + (size_t)srcN * NN + part * 8);
      if (part < 8) {   // e = ea@W2 + b2 (K=2) -> cols 256..319
        const float a0 = ea[eg * 2 + 0], a1 = ea[eg * 2 + 1];
#pragma unroll
        for (int j = 0; j < 8; ++j) {
          const int c = part * 8 + j;
          zs[eL * ZPAD + 2 * NN + c] =
              (__bf16)(fmaf(a0, W2[c], fmaf(a1, W2[EN + c], b2[c])));
        }
      }
    }
    __syncthreads();

    // ---- WMMA: A from LDS (16x32 per step, double-buffered), B in registers
    v8f accf = {}; v8f accs = {};
    const __bf16* arow = zs + lm * ZPAD;
    union AV { v16bf v; uint4 q[2]; };
    AV a0, a1;
    // A k order chosen here (matches B): lane group `half` holds contiguous K[k0+16*half .. +15]
    a0.q[0] = *(const uint4*)(arow + half * 16);
    a0.q[1] = *(const uint4*)(arow + half * 16 + 8);
#pragma unroll
    for (int s = 0; s < KSTEPS; ++s) {
      if (s + 1 < KSTEPS) {
        const int k1 = (s + 1) * 32;
        a1.q[0] = *(const uint4*)(arow + k1 + half * 16);
        a1.q[1] = *(const uint4*)(arow + k1 + half * 16 + 8);
      }
      accf = __builtin_amdgcn_wmma_f32_16x16x32_bf16(false, a0.v, false, Bf[s],
                                                     (short)0, accf, false, false);
      accs = __builtin_amdgcn_wmma_f32_16x16x32_bf16(false, a0.v, false, Bs[s],
                                                     (short)0, accs, false, false);
      if (s + 1 < KSTEPS) a0 = a1;
    }

    // ---- gate + scatter-add (D layout: VGPR r -> row r+8*half, col = lane%16)
#pragma unroll
    for (int r = 0; r < 8; ++r) {
      const int m = r + half * 8;
      const float f = accf[r] + bgate;
      const float s = accs[r] + bsoft;
      const float sig = 1.f / (1.f + __expf(-f));
      // branchless stable softplus: max(s,0) + log1p(exp(-|s|))
      const float sp = fmaxf(s, 0.f) + log1pf(__expf(-fabsf(s)));
      atomicAdd(acc + (size_t)dsts[m] * NN + col, sig * sp);
    }
  }
}

// ------------------------------------- h = tanh(h + acc); refresh bf16; acc=0
__global__ void k_resid_tanh(float* __restrict__ h, __bf16* __restrict__ hbf,
                             float* __restrict__ acc, int n) {
  int i = blockIdx.x * 256 + threadIdx.x;
  if (i >= n) return;
  float v = tanhf(h[i] + acc[i]);
  h[i] = v; hbf[i] = (__bf16)v; acc[i] = 0.f;
}

// ----------------------------------------------------- global_add_pool via atomics
__global__ void k_pool(const float* __restrict__ h, const int* __restrict__ batch,
                       float* __restrict__ g, int n) {
  int i = blockIdx.x * 256 + threadIdx.x;
  if (i >= n) return;
  int node = i >> 7, c = i & (NN - 1);
  atomicAdd(g + (size_t)batch[node] * NN + c, h[i]);
}

// ----------------------------------------- small dense: out = in@W + b  (G rows)
__global__ void k_lin(const float* __restrict__ in, const float* __restrict__ W,
                      const float* __restrict__ b, float* __restrict__ out,
                      int K, int Nc) {
  int r = blockIdx.y;
  int c = blockIdx.x * 64 + threadIdx.x;
  if (c >= Nc) return;
  float a = b[c];
  for (int k = 0; k < K; ++k) a = fmaf(in[r * K + k], W[k * Nc + c], a);
  out[r * Nc + c] = a;
}

// ----------------- training-mode BN over G=512 rows (+ optional ReLU), in place
__global__ void __launch_bounds__(256)
k_bn_act(float* __restrict__ y, const float* __restrict__ g,
         const float* __restrict__ b, int Nc, int relu) {
  const int c = blockIdx.x;
  __shared__ float sm[256], sv[256];
  float s = 0.f, ss = 0.f;
  for (int r = threadIdx.x; r < kGraphs; r += 256) {
    float v = y[r * Nc + c]; s += v; ss += v * v;
  }
  sm[threadIdx.x] = s; sv[threadIdx.x] = ss;
  __syncthreads();
  for (int st = 128; st > 0; st >>= 1) {
    if (threadIdx.x < st) { sm[threadIdx.x] += sm[threadIdx.x + st];
                            sv[threadIdx.x] += sv[threadIdx.x + st]; }
    __syncthreads();
  }
  const float m   = sm[0] / (float)kGraphs;
  const float var = sv[0] / (float)kGraphs - m * m;
  const float inv = rsqrtf(var + EPS) * g[c];
  for (int r = threadIdx.x; r < kGraphs; r += 256) {
    float v = (y[r * Nc + c] - m) * inv + b[c];
    if (relu) v = fmaxf(v, 0.f);
    y[r * Nc + c] = v;
  }
}

// ------------------------------------------ head output: sigmoid(in@Wsig + b)
__global__ void k_head_out(const float* __restrict__ in, const float* __restrict__ W,
                           const float* __restrict__ b, float* __restrict__ out) {
  int r = blockIdx.x * 256 + threadIdx.x;
  if (r >= kGraphs) return;
  float a = b[0];
#pragma unroll
  for (int k = 0; k < 10; ++k) a = fmaf(in[r * 10 + k], W[k], a);
  out[r] = 1.f / (1.f + __expf(-a));
}

// ---------------------------------------------------------------------------
static inline char* ws_take(char*& p, size_t bytes) {
  char* r = p;
  p += (bytes + 255) & ~(size_t)255;
  return r;
}

extern "C" void kernel_launch(void* const* d_in, const int* in_sizes, int n_in,
                              void* d_out, int out_size, void* d_ws, size_t ws_size,
                              hipStream_t stream) {
  (void)in_sizes; (void)n_in; (void)out_size; (void)ws_size;
  // inputs (setup_inputs order; params flattened in dict insertion order)
  const float* x     = (const float*)d_in[0];
  const int*   ei    = (const int*)d_in[1];      // [2, E]
  const float* ea    = (const float*)d_in[2];    // [E, 2]
  const int*   batch = (const int*)d_in[3];
  int p = 4;
  auto F = [&](void) { return (const float*)d_in[p++]; };
  const float *g1W = F(), *g1b = F();                      // gnn1
  const float *g2W = F(), *g2b = F();                      // gnn2
  const float *c1Wf = F(), *c1bf = F(), *c1Ws = F(), *c1bs = F();  // conv1
  const float *c2Wf = F(), *c2bf = F(), *c2Ws = F(), *c2bs = F();  // conv2
  const float *l1W = F(), *l1b = F(), *bn1g = F(), *bn1b = F();
  const float *l2W = F(), *l2b = F(), *bn2g = F(), *bn2b = F();
  const float *l3W = F(), *l3b = F(), *bn3g = F(), *bn3b = F();
  const float *l4W = F(), *l4b = F(), *bn4g = F(), *bn4b = F();
  const float* ts[5][10];
  for (int i = 0; i < 5; ++i)
    for (int j = 0; j < 10; ++j) ts[i][j] = F();
  float* out = (float*)d_out;

  // workspace
  char* wp = (char*)d_ws;
  float*  h    = (float*)ws_take(wp, (size_t)kNodes * NN * 4);
  __bf16* hbf  = (__bf16*)ws_take(wp, (size_t)kNodes * NN * 2);
  float*  accb = (float*)ws_take(wp, (size_t)kNodes * NN * 4);
  __bf16* wtf1 = (__bf16*)ws_take(wp, (size_t)NN * ZDIM * 2);
  __bf16* wts1 = (__bf16*)ws_take(wp, (size_t)NN * ZDIM * 2);
  __bf16* wtf2 = (__bf16*)ws_take(wp, (size_t)NN * ZDIM * 2);
  __bf16* wts2 = (__bf16*)ws_take(wp, (size_t)NN * ZDIM * 2);
  float*  gbuf = (float*)ws_take(wp, (size_t)kGraphs * NN * 4);
  float*  bufA = (float*)ws_take(wp, (size_t)kGraphs * 200 * 4);
  float*  bufB = (float*)ws_take(wp, (size_t)kGraphs * 200 * 4);
  float*  bufC = (float*)ws_take(wp, (size_t)kGraphs * 200 * 4);

  const int nh = kNodes * NN;                       // 6.4M

  // 0) zero accumulators (every launch — harness does not re-poison)
  k_zero<<<(nh + 255) / 256, 256, 0, stream>>>(accb, nh);
  k_zero<<<(kGraphs * NN + 255) / 256, 256, 0, stream>>>(gbuf, kGraphs * NN);

  // 1) embeds + weight prep
  k_node_embed<<<kNodes, 128, 0, stream>>>(x, g1W, g1b, h, hbf);
  const int wtn = (ZDIM * NN + 255) / 256;
  k_prep_wt<<<wtn, 256, 0, stream>>>(c1Wf, wtf1);
  k_prep_wt<<<wtn, 256, 0, stream>>>(c1Ws, wts1);
  k_prep_wt<<<wtn, 256, 0, stream>>>(c2Wf, wtf2);
  k_prep_wt<<<wtn, 256, 0, stream>>>(c2Ws, wts2);

  // 2) conv1 + tanh ; conv2 + tanh
  k_cgconv<<<kConvBlocks, 256, 0, stream>>>(hbf, ei, ea, g2W, g2b,
                                            wtf1, c1bf, wts1, c1bs, accb);
  k_resid_tanh<<<(nh + 255) / 256, 256, 0, stream>>>(h, hbf, accb, nh);
  k_cgconv<<<kConvBlocks, 256, 0, stream>>>(hbf, ei, ea, g2W, g2b,
                                            wtf2, c2bf, wts2, c2bs, accb);
  k_resid_tanh<<<(nh + 255) / 256, 256, 0, stream>>>(h, hbf, accb, nh);

  // 3) global add pool
  k_pool<<<(nh + 255) / 256, 256, 0, stream>>>(h, batch, gbuf, nh);

  // 4) trunk: 128->200->100->50->10 with BN+ReLU
  k_lin<<<dim3((200 + 63) / 64, kGraphs), 64, 0, stream>>>(gbuf, l1W, l1b, bufA, NN, 200);
  k_bn_act<<<200, 256, 0, stream>>>(bufA, bn1g, bn1b, 200, 1);
  k_lin<<<dim3((100 + 63) / 64, kGraphs), 64, 0, stream>>>(bufA, l2W, l2b, bufB, 200, 100);
  k_bn_act<<<100, 256, 0, stream>>>(bufB, bn2g, bn2b, 100, 1);
  k_lin<<<dim3(1, kGraphs), 64, 0, stream>>>(bufB, l3W, l3b, bufA, 100, 50);
  k_bn_act<<<50, 256, 0, stream>>>(bufA, bn3g, bn3b, 50, 1);
  k_lin<<<dim3(1, kGraphs), 64, 0, stream>>>(bufA, l4W, l4b, bufB, 50, 10);
  k_bn_act<<<10, 256, 0, stream>>>(bufB, bn4g, bn4b, 10, 1);

  // 5) five task heads: 10->50 (BN,ReLU) ->10 (BN,ReLU) ->1 sigmoid
  for (int i = 0; i < 5; ++i) {
    k_lin<<<dim3(1, kGraphs), 64, 0, stream>>>(bufB, ts[i][0], ts[i][1], bufC, 10, 50);
    k_bn_act<<<50, 256, 0, stream>>>(bufC, ts[i][2], ts[i][3], 50, 1);
    k_lin<<<dim3(1, kGraphs), 64, 0, stream>>>(bufC, ts[i][4], ts[i][5], bufA, 50, 10);
    k_bn_act<<<10, 256, 0, stream>>>(bufA, ts[i][6], ts[i][7], 10, 1);
    k_head_out<<<2, 256, 0, stream>>>(bufA, ts[i][8], ts[i][9], out + (size_t)i * kGraphs);
  }
}
